// LSTMVAE_25786983645771
// MI455X (gfx1250) — matrix operand order repaired and checked by reading, not compiled, not run on hardware
//
#include <hip/hip_runtime.h>
#include <hip/hip_bf16.h>
#include <math.h>

typedef __bf16 bf16_t;
typedef __attribute__((ext_vector_type(16))) __bf16 v16bf;
typedef __attribute__((ext_vector_type(8)))  float  v8f;

#define B_ 32
#define T_ 512
#define H_ 1024
#define L_ 256
#define V_ 13
#define HL_ (H_ + L_)   // 1280
#define UPITCH 1032     // bf16 pitch: 2064B = 516 dwords, 516 % 64 = 4 -> banks spread
#define NBLK 32         // persistent recurrence workgroups (co-resident on MI455X)

// fast activations: single v_exp_f32 + v_rcp_f32 instead of IEEE divide chains
__device__ __forceinline__ float sigmoidf_(float x) {
  return __builtin_amdgcn_rcpf(1.0f + __expf(-x));     // x->-inf: rcp(inf)=0; x->+inf: 1
}
__device__ __forceinline__ float tanhf_(float x) {
  return 1.0f - 2.0f * __builtin_amdgcn_rcpf(1.0f + __expf(2.0f * x)); // saturates to +/-1
}

// ---------------- prep: f32 -> bf16 ----------------
__global__ void cvt_bf16_kernel(const float* __restrict__ src, bf16_t* __restrict__ dst, int n) {
  int i = blockIdx.x * blockDim.x + threadIdx.x;
  if (i < n) dst[i] = (bf16_t)src[i];
}

// ---------------- h0 = relu(latent @ Wz^T + bz), c0 = 0; also reset grid barrier ----
__global__ void h0_kernel(const float* __restrict__ latent, const float* __restrict__ Wz,
                          const float* __restrict__ bz, bf16_t* __restrict__ h_bf,
                          float* __restrict__ c, int* __restrict__ cnt, int* __restrict__ gen) {
  int i = blockIdx.x * blockDim.x + threadIdx.x;       // [0, B*H)
  if (i == 0) { *cnt = 0; *gen = 0; }
  int b = i >> 10, col = i & (H_ - 1);
  const float* lv = latent + b * L_;
  const float* wr = Wz + col * L_;
  float s = bz[col];
  for (int l = 0; l < L_; ++l) s += lv[l] * wr[l];
  h_bf[i] = (bf16_t)fmaxf(s, 0.0f);
  c[i] = 0.0f;
}

// ---------------- persistent LSTM recurrence: all 512 steps in one kernel ----------
// grid: 32 WGs x 256 threads (8 waves). WG owns h-columns [32*bx, 32*bx+32).
// wave w: gate = w>>1 (i,f,g,o), ct = w&1 (col subtile). Each wave computes BOTH
// batch M-tiles (rows 0-15 and 16-31) reusing one B tile -> 2 wmma per K-iter.
__global__ __launch_bounds__(256) void lstm_persistent_kernel(
    const float* __restrict__ seq_input, const bf16_t* __restrict__ Whh_bf,
    const float* __restrict__ W_ih, const float* __restrict__ b_ih,
    const float* __restrict__ b_hh, float* __restrict__ c,
    bf16_t* __restrict__ h0buf, bf16_t* __restrict__ h1buf,
    bf16_t* __restrict__ hs, int* __restrict__ cnt, int* __restrict__ gen) {
  __shared__ float gbuf[4][B_][32];     // 16 KB gate exchange
  __shared__ float xbuf[B_][16];        // x_t staging (13 used)
  __shared__ float wihL[4][32][14];     // W_ih slice for this WG's columns
  __shared__ float biasL[4][32];        // b_ih + b_hh fused

  const int tid  = threadIdx.x;
  const int lane = tid & 31, wave = tid >> 5;
  const int gate = wave >> 1, ct = wave & 1;
  const int half = lane >> 4, ln = lane & 15, half8 = half * 8;
  const int colbase = blockIdx.x * 32;
  const int cl   = ct * 16 + ln;                       // local col 0..31
  const int jcol = gate * H_ + colbase + cl;           // W_hh row / gate column

  // one-time: stage W_ih slice + fused bias into LDS
  for (int idx = tid; idx < 4 * 32 * V_; idx += 256) {
    int g = idx / (32 * V_), r = idx - g * (32 * V_);
    int ccl = r / V_, v = r - ccl * V_;
    wihL[g][ccl][v] = W_ih[(size_t)(g * H_ + colbase + ccl) * V_ + v];
  }
  if (tid < 128) {
    int g = tid >> 5, ccl = tid & 31;
    int j = g * H_ + colbase + ccl;
    biasL[g][ccl] = b_ih[j] + b_hh[j];
  }
  __syncthreads();

  const bf16_t* wrow = Whh_bf + (size_t)jcol * H_;

  for (int t = 0; t < T_; ++t) {
    const bf16_t* h_in  = (t & 1) ? h1buf : h0buf;
    bf16_t*       h_out = (t & 1) ? h0buf : h1buf;

    // stage x_t into LDS (+ prefetch x_{t+1})
    for (int idx = tid; idx < B_ * V_; idx += 256) {
      int b = idx / V_, v = idx - b * V_;
      const float* xp = seq_input + ((size_t)b * T_ + t) * V_;
      xbuf[b][v] = xp[v];
      if (t + 1 < T_) __builtin_prefetch(xp + V_, 0, 1);
    }
    __syncthreads();

    // C init = x_t . W_ih[j,:] + b_ih[j] + b_hh[j]  (LDS reads only)
    v8f acc0, acc1;
    {
      const float bias = biasL[gate][cl];
#pragma unroll
      for (int r = 0; r < 8; ++r) {
        int m0 = half8 + r, m1 = 16 + half8 + r;       // matches C/D layout rows
        float s0 = bias, s1 = bias;
#pragma unroll
        for (int v = 0; v < V_; ++v) {
          float w = wihL[gate][cl][v];
          s0 += xbuf[m0][v] * w;
          s1 += xbuf[m1][v] * w;
        }
        acc0[r] = s0; acc1[r] = s1;
      }
    }

    // gates += h @ W_hh^T : v_wmma_f32_16x16x32_bf16, K = 1024, B reused for 2 M-tiles
    for (int k0 = 0; k0 < H_; k0 += 32) {
      v16bf a0, a1, bb;
#pragma unroll
      for (int e = 0; e < 16; ++e) {                   // A 16x32 bf16 layout
        int ka = k0 + ((e & 8) ? 16 : 0) + half8 + (e & 7);
        a0[e] = h_in[ln * H_ + ka];
        a1[e] = h_in[(16 + ln) * H_ + ka];
      }
#pragma unroll
      for (int e = 0; e < 16; ++e)                     // B 32x16: K = (e&8?16:0) + half*8 + (e&7), N = ln
        bb[e] = wrow[k0 + ((e & 8) ? 16 : 0) + half8 + (e & 7)];
      acc0 = __builtin_amdgcn_wmma_f32_16x16x32_bf16(false, a0, false, bb, (short)0, acc0,
                                                     false, false);
      acc1 = __builtin_amdgcn_wmma_f32_16x16x32_bf16(false, a1, false, bb, (short)0, acc1,
                                                     false, false);
    }

#pragma unroll
    for (int r = 0; r < 8; ++r) {
      gbuf[gate][half8 + r][cl]      = acc0[r];
      gbuf[gate][16 + half8 + r][cl] = acc1[r];
    }
    __syncthreads();

    // activations + state update: 32 batches x 32 columns, 4 per thread
    for (int idx = tid; idx < B_ * 32; idx += 256) {
      int b = idx >> 5, ccl = idx & 31;
      int col = colbase + ccl;
      float ig = sigmoidf_(gbuf[0][b][ccl]);
      float fg = sigmoidf_(gbuf[1][b][ccl]);
      float gg = tanhf_(gbuf[2][b][ccl]);
      float og = sigmoidf_(gbuf[3][b][ccl]);
      float cv = fg * c[b * H_ + col] + ig * gg;
      float hv = og * tanhf_(cv);
      c[b * H_ + col] = cv;
      bf16_t hb = (bf16_t)hv;
      h_out[b * H_ + col] = hb;                        // ping-pong for next step
      hs[((size_t)b * T_ + t) * H_ + col] = hb;        // history for the head
    }

    // device-wide sense-reversing barrier between steps.
    // Every wave fences so each CU's WGP$ is written back / invalidated.
    __threadfence();
    __syncthreads();
    if (tid == 0) {
      int g = __hip_atomic_load(gen, __ATOMIC_RELAXED, __HIP_MEMORY_SCOPE_AGENT);
      if (__hip_atomic_fetch_add(cnt, 1, __ATOMIC_ACQ_REL, __HIP_MEMORY_SCOPE_AGENT) ==
          NBLK - 1) {
        __hip_atomic_store(cnt, 0, __ATOMIC_RELAXED, __HIP_MEMORY_SCOPE_AGENT);
        __hip_atomic_fetch_add(gen, 1, __ATOMIC_RELEASE, __HIP_MEMORY_SCOPE_AGENT);
      } else {
        while (__hip_atomic_load(gen, __ATOMIC_ACQUIRE, __HIP_MEMORY_SCOPE_AGENT) == g)
          __builtin_amdgcn_s_sleep(1);
      }
    }
    __syncthreads();
    __threadfence();
  }
}

// ---------------- output head ----------------
__global__ __launch_bounds__(256) void head_kernel(
    const bf16_t* __restrict__ hs, const bf16_t* __restrict__ z_bf,
    const bf16_t* __restrict__ Wnl_bf, const float* __restrict__ bnl,
    const float* __restrict__ Wout, const float* __restrict__ bout,
    const int* __restrict__ labels, float* __restrict__ nll) {
  __shared__ bf16_t u[16][UPITCH];                     // ~33 KB
  __shared__ float lg[16][16];                         // 13 used
  const int lane  = threadIdx.x & 31;
  const int wave  = threadIdx.x >> 5;
  const int half  = lane >> 4;
  const int ln    = lane & 15;
  const int half8 = half * 8;
  const int row0  = blockIdx.x * 16;

  const int arow = row0 + ln;                          // A-row = lane&15
  const int ab = arow >> 9, at = arow & (T_ - 1);
  const bf16_t* hrow = hs + ((size_t)ab * T_ + at) * H_;
  const bf16_t* zrow = z_bf + ab * L_;

  for (int pass = 0; pass < 8; ++pass) {
    const int ncol = pass * 128 + wave * 16;
    const bf16_t* wn = Wnl_bf + (size_t)(ncol + ln) * HL_;
    v8f acc = {};
    for (int k0 = 0; k0 < HL_; k0 += 32) {
      v16bf a, bb;
#pragma unroll
      for (int e = 0; e < 16; ++e) {
        int ka = k0 + ((e & 8) ? 16 : 0) + half8 + (e & 7);
        a[e] = (ka < H_) ? hrow[ka] : zrow[ka - H_];   // feat = [h_t , z]
      }
#pragma unroll
      for (int e = 0; e < 16; ++e) bb[e] = wn[k0 + ((e & 8) ? 16 : 0) + half8 + (e & 7)];
      acc = __builtin_amdgcn_wmma_f32_16x16x32_bf16(false, a, false, bb, (short)0, acc,
                                                    false, false);
    }
    const int n = ncol + ln;
    const float bn = bnl[n];
#pragma unroll
    for (int r = 0; r < 8; ++r)
      u[half8 + r][n] = (bf16_t)fmaxf(acc[r] + bn, 0.0f);
  }
  __syncthreads();

  // logits[16][13]: K=1024 dot per element
  for (int idx = threadIdx.x; idx < 16 * V_; idx += 256) {
    int m = idx / V_, v = idx - m * V_;
    const float* wo = Wout + v * H_;
    float s = bout[v];
    for (int k = 0; k < H_; ++k) s += (float)u[m][k] * wo[k];
    lg[m][v] = s;
  }
  __syncthreads();

  if (threadIdx.x < 16) {
    int m = threadIdx.x;
    int row = row0 + m;
    int b = row >> 9, t = row & (T_ - 1);
    float mx = lg[m][0];
#pragma unroll
    for (int v = 1; v < V_; ++v) mx = fmaxf(mx, lg[m][v]);
    float se = 0.0f;
#pragma unroll
    for (int v = 0; v < V_; ++v) se += __expf(lg[m][v] - mx);
    float lse = mx + __logf(se);
    int lbl = labels[b * T_ + t];
    nll[b * T_ + t] = lse - lg[m][lbl];
  }
}

// ---------------- masked segment sum: one wave per batch row ----------------
__global__ __launch_bounds__(1024) void reduce_kernel(
    const float* __restrict__ nll, const int* __restrict__ lengths,
    float* __restrict__ out) {
  __shared__ float sums[B_];
  const int lane = threadIdx.x & 31;
  const int b = threadIdx.x >> 5;
  const int len = lengths[b];
  float s = 0.0f;
  for (int t = lane; t < T_; t += 32)
    if (t < len) s += nll[b * T_ + t];
#pragma unroll
  for (int o = 16; o > 0; o >>= 1) s += __shfl_xor(s, o, 32);
  if (lane == 0) sums[b] = s;
  __syncthreads();
  if (threadIdx.x == 0) {
    float tot = 0.0f;
    for (int i = 0; i < B_; ++i) tot += sums[i];
    out[0] = tot;
  }
  if (threadIdx.x < B_) out[1 + threadIdx.x] = sums[threadIdx.x];
}

extern "C" void kernel_launch(void* const* d_in, const int* in_sizes, int n_in,
                              void* d_out, int out_size, void* d_ws, size_t ws_size,
                              hipStream_t stream) {
  (void)in_sizes; (void)n_in; (void)out_size; (void)ws_size;
  const float* seq_input = (const float*)d_in[0];
  const float* latent    = (const float*)d_in[1];
  const int*   labels    = (const int*)d_in[2];
  const int*   lengths   = (const int*)d_in[3];
  const float* W_ih = (const float*)d_in[4];
  const float* W_hh = (const float*)d_in[5];
  const float* b_ih = (const float*)d_in[6];
  const float* b_hh = (const float*)d_in[7];
  const float* Wz   = (const float*)d_in[8];
  const float* bz   = (const float*)d_in[9];
  const float* Wnl  = (const float*)d_in[10];
  const float* bnl  = (const float*)d_in[11];
  const float* Wout = (const float*)d_in[12];
  const float* bout = (const float*)d_in[13];
  float* out = (float*)d_out;

  char* ws = (char*)d_ws;
  size_t off = 0;
  auto alloc = [&](size_t bytes) -> void* {
    void* p = ws + off;
    off = (off + bytes + 255) & ~(size_t)255;
    return p;
  };
  bf16_t* Whh_bf = (bf16_t*)alloc((size_t)4 * H_ * H_ * sizeof(bf16_t));   // 8 MB
  bf16_t* Wnl_bf = (bf16_t*)alloc((size_t)H_ * HL_ * sizeof(bf16_t));      // 2.5 MB
  bf16_t* z_bf   = (bf16_t*)alloc((size_t)B_ * L_ * sizeof(bf16_t));
  float*  c      = (float*) alloc((size_t)B_ * H_ * sizeof(float));
  bf16_t* hb0    = (bf16_t*)alloc((size_t)B_ * H_ * sizeof(bf16_t));
  bf16_t* hb1    = (bf16_t*)alloc((size_t)B_ * H_ * sizeof(bf16_t));
  bf16_t* hs     = (bf16_t*)alloc((size_t)B_ * T_ * H_ * sizeof(bf16_t));  // 32 MB
  float*  nll    = (float*) alloc((size_t)B_ * T_ * sizeof(float));
  int*    barrier_ws = (int*)alloc(2 * sizeof(int));
  int* cnt = barrier_ws;
  int* gen = barrier_ws + 1;

  int n1 = 4 * H_ * H_;
  cvt_bf16_kernel<<<(n1 + 255) / 256, 256, 0, stream>>>(W_hh, Whh_bf, n1);
  int n2 = H_ * HL_;
  cvt_bf16_kernel<<<(n2 + 255) / 256, 256, 0, stream>>>(Wnl, Wnl_bf, n2);
  int n3 = B_ * L_;
  cvt_bf16_kernel<<<(n3 + 255) / 256, 256, 0, stream>>>(latent, z_bf, n3);

  h0_kernel<<<(B_ * H_) / 256, 256, 0, stream>>>(latent, Wz, bz, hb0, c, cnt, gen);

  lstm_persistent_kernel<<<NBLK, 256, 0, stream>>>(seq_input, Whh_bf, W_ih, b_ih, b_hh,
                                                   c, hb0, hb1, hs, cnt, gen);

  head_kernel<<<(B_ * T_) / 16, 256, 0, stream>>>(hs, z_bf, Wnl_bf, bnl, Wout, bout,
                                                  labels, nll);
  reduce_kernel<<<1, 1024, 0, stream>>>(nll, lengths, out);
}
